// Decoder_26087631356046
// MI455X (gfx1250) — compile-verified, matrix operand order checked by reading, not run on hardware
//
#include <hip/hip_runtime.h>
#include <hip/hip_bf16.h>

// ---------------------------------------------------------------------------
// Problem dimensions (fixed by the reference)
// ---------------------------------------------------------------------------
#define B_   32
#define L_   512
#define X_   64
#define HH_  256      // H/2
#define H2_  1024     // 2H
#define H4_  2048     // 4H
#define LP_  (L_ + 2) // padded length for conv
#define M_   (B_ * L_)
#define NLAYERS_ 5

typedef _Float16 v16h __attribute__((ext_vector_type(16)));
typedef _Float16 v8h  __attribute__((ext_vector_type(8)));
typedef _Float16 v4h  __attribute__((ext_vector_type(4)));
typedef float    v8f  __attribute__((ext_vector_type(8)));

#define TK   32
#define LSTR 40   // LDS row stride in halves (32 + 8 pad); 80B => keeps 16B chunk alignment

// CDNA5 async global->LDS copy (ASYNCcnt path, CDNA5 ISA).
// lds_off = byte offset within LDS (low 32 bits of a generic pointer to
// __shared__), gaddr = 64-bit global address. Moves 16 bytes per lane.
__device__ __forceinline__ void async_ld_b128(unsigned lds_off, unsigned long long gaddr) {
  asm volatile("global_load_async_to_lds_b128 %0, %1, off"
               :: "v"(lds_off), "v"(gaddr) : "memory");
}
__device__ __forceinline__ void wait_async0() {
  asm volatile("s_wait_asynccnt 0x0" ::: "memory");
}

__device__ __forceinline__ v16h frag_ld(const _Float16* base) {
  // One 16x32 f16 fragment per CDNA5 16-bit A/B layout:
  //   lanes 0-15 : row = lane,     K in {0..7} U {16..23}
  //   lanes 16-31: row = lane-16,  K in {8..15} U {24..31}
  int lane = threadIdx.x & 31;
  int r = lane & 15;
  int kb = (lane < 16) ? 0 : 8;
  const _Float16* p = base + r * LSTR + kb;
  v16h v;
#pragma unroll
  for (int i = 0; i < 8; ++i) { v[i] = p[i]; v[i + 8] = p[16 + i]; }
  return v;
}

// ---------------------------------------------------------------------------
// Generic WMMA GEMM:  C = A(f32, cast->f16) @ W(f16)^T [+ bias] [+ addsrc]
//   A row r read at:   A + a_off + r*lda + (r/rpb)*a_be + k
//   W row n read at:   W + (r/rpb)*w_bs + n*ldw + k
//   C (r,n) at:        C + c_off + r*ldc + (r/rpb)*c_be + n
//   addsrc optional, contiguous [M, ldc].
// Template: wave tile = (WMT*16) x (WNT*16); block = WVM x WVN waves.
// Double-buffered LDS; B tile optionally via async global->LDS.
// Requires M % BM == 0, N % BN == 0, K % 32 == 0, rpb % BM == 0,
// all row strides multiples of 8 halves / 4 floats, 16B-aligned bases.
// ---------------------------------------------------------------------------
template <int WMT, int WNT, int WVM, int WVN, bool ASYNC_B>
__global__ __launch_bounds__(WVM * WVN * 32)
void wmma_gemm_kernel(const float* __restrict__ A, long long a_off, int lda, long long a_be,
                      const _Float16* __restrict__ W, int ldw, long long w_bs,
                      const float* __restrict__ bias,
                      const float* __restrict__ addsrc,
                      float* __restrict__ C, long long c_off, int ldc, long long c_be,
                      int M, int N, int K, int rpb) {
  constexpr int BM = WVM * WMT * 16;
  constexpr int BN = WVN * WNT * 16;
  constexpr int NT = WVM * WVN * 32;

  __shared__ _Float16 sA[2][BM * LSTR];
  __shared__ _Float16 sB[2][BN * LSTR];

  const int t = threadIdx.x;
  const int lane = t & 31;
  const int wave = t >> 5;
  const int wm = wave % WVM;
  const int wn = wave / WVM;

  const int rowBase = blockIdx.y * BM;
  const int colBase = blockIdx.x * BN;
  const int batch = rowBase / rpb;

  const float* Ab = A + a_off + (long long)batch * a_be + (long long)rowBase * lda;
  const _Float16* Wb = W + (long long)batch * w_bs + (long long)colBase * ldw;

  v8f acc[WMT][WNT];
#pragma unroll
  for (int mi = 0; mi < WMT; ++mi)
#pragma unroll
    for (int ni = 0; ni < WNT; ++ni)
#pragma unroll
      for (int e = 0; e < 8; ++e) acc[mi][ni][e] = 0.0f;

  // ---- tile loaders --------------------------------------------------------
  auto load_tile = [&](int buf, int kb) {
    // A: BM x 32 f32 -> f16, float4 chunks (BM*8 chunks)
#pragma unroll
    for (int ch = t; ch < BM * (TK / 4); ch += NT) {
      int r = ch >> 3;
      int c = (ch & 7) * 4;
      const float4 f = *(const float4*)(Ab + (long long)r * lda + kb + c);
      v4h hv;
      hv[0] = (_Float16)f.x; hv[1] = (_Float16)f.y;
      hv[2] = (_Float16)f.z; hv[3] = (_Float16)f.w;
      *(v4h*)&sA[buf][r * LSTR + c] = hv;
    }
    // B: BN x 32 f16, 16-byte chunks (BN*4 chunks)
#pragma unroll
    for (int ch = t; ch < BN * (TK / 8); ch += NT) {
      int r = ch >> 2;
      int c = (ch & 3) * 8;
      const _Float16* g = Wb + (long long)r * ldw + kb + c;
      if (ASYNC_B) {
        unsigned loff = (unsigned)(unsigned long long)(const void*)&sB[buf][r * LSTR + c];
        async_ld_b128(loff, (unsigned long long)g);
      } else {
        *(v8h*)&sB[buf][r * LSTR + c] = *(const v8h*)g;
      }
    }
  };

  auto prefetch_tile = [&](int kb) {
    // one speculative cacheline prefetch per thread for A and B of tile kb
    int ra = t >> 3, ca = (t & 7) * 4;
    __builtin_prefetch(Ab + (long long)ra * lda + kb + ca, 0, 1);
    int rb = t >> 2, cb = (t & 3) * 8;
    __builtin_prefetch(Wb + (long long)rb * ldw + kb + cb, 0, 1);
  };

  auto compute = [&](int buf) {
    v16h af[WMT], bf[WNT];
#pragma unroll
    for (int mi = 0; mi < WMT; ++mi)
      af[mi] = frag_ld(&sA[buf][(wm * WMT * 16 + mi * 16) * LSTR]);
#pragma unroll
    for (int ni = 0; ni < WNT; ++ni)
      bf[ni] = frag_ld(&sB[buf][(wn * WNT * 16 + ni * 16) * LSTR]);
#pragma unroll
    for (int mi = 0; mi < WMT; ++mi)
#pragma unroll
      for (int ni = 0; ni < WNT; ++ni)
        acc[mi][ni] = __builtin_amdgcn_wmma_f32_16x16x32_f16(
            false, af[mi], false, bf[ni], (short)0, acc[mi][ni], false, false);
  };

  // ---- main loop: double-buffered ------------------------------------------
  load_tile(0, 0);
  if (ASYNC_B) wait_async0();
  __syncthreads();

  int cur = 0;
  for (int kb = 0; kb < K; kb += TK) {
    const int nxt = kb + TK;
    if (nxt < K) {
      load_tile(cur ^ 1, nxt);
      if (nxt + TK < K) prefetch_tile(nxt + TK);
    }
    compute(cur);
    if (ASYNC_B && nxt < K) wait_async0();
    __syncthreads();
    cur ^= 1;
  }

  // ---- epilogue ------------------------------------------------------------
  // C/D layout: lanes 0-15 -> N=lane, M=VGPRj; lanes 16-31 -> N=lane-16, M=j+8.
  float* Cp = C + c_off + (long long)batch * c_be;
  const int nIn = lane & 15;
  const int mOfs = (lane < 16) ? 0 : 8;
  if (addsrc) {
#pragma unroll
    for (int mi = 0; mi < WMT; ++mi)
#pragma unroll
      for (int ni = 0; ni < WNT; ++ni) {
        int n = colBase + wn * WNT * 16 + ni * 16 + nIn;
        float bv = bias ? bias[n] : 0.0f;
        v8f a = acc[mi][ni];
#pragma unroll
        for (int j = 0; j < 8; ++j) {
          long long row = rowBase + wm * WMT * 16 + mi * 16 + mOfs + j;
          Cp[row * ldc + n] = a[j] + bv + addsrc[row * ldc + n];
        }
      }
  } else {
#pragma unroll
    for (int mi = 0; mi < WMT; ++mi)
#pragma unroll
      for (int ni = 0; ni < WNT; ++ni) {
        int n = colBase + wn * WNT * 16 + ni * 16 + nIn;
        float bv = bias ? bias[n] : 0.0f;
        v8f a = acc[mi][ni];
#pragma unroll
        for (int j = 0; j < 8; ++j) {
          long long row = rowBase + wm * WMT * 16 + mi * 16 + mOfs + j;
          Cp[row * ldc + n] = a[j] + bv;
        }
      }
  }
}

// ---------------------------------------------------------------------------
// Small helper kernels
// ---------------------------------------------------------------------------
__global__ void cvt_f16_kernel(const float* __restrict__ in, _Float16* __restrict__ out,
                               long long n) {
  long long i = (long long)blockIdx.x * blockDim.x + threadIdx.x;
  if (i < n) out[i] = (_Float16)in[i];
}

// conv_w [5][H4][H2][3] f32 -> per-layer f16 [H4][3*H2], k = t*H2 + i
__global__ void conv_repack_kernel(const float* __restrict__ in, _Float16* __restrict__ out,
                                   long long n) {
  long long idx = (long long)blockIdx.x * blockDim.x + threadIdx.x;
  if (idx >= n) return;
  int i = (int)(idx % H2_);
  long long r = idx / H2_;
  int tt = (int)(r % 3);
  r /= 3;
  int o = (int)(r % H4_);
  int l = (int)(r / H4_);
  out[idx] = (_Float16)in[(((long long)l * H4_ + o) * H2_ + i) * 3 + tt];
}

// residual_encoder [B][L][HH] f32 -> f16 transposed [B][HH][L]
__global__ void transpose_re_kernel(const float* __restrict__ in, _Float16* __restrict__ out) {
  long long idx = (long long)blockIdx.x * blockDim.x + threadIdx.x;
  if (idx >= (long long)B_ * HH_ * L_) return;
  int m = (int)(idx % L_);
  long long r = idx / L_;
  int d = (int)(r % HH_);
  int b = (int)(r / HH_);
  out[idx] = (_Float16)in[((long long)b * L_ + m) * HH_ + d];
}

// bias_emb[h] = lin_b[h] + pos_b[h] + sum_l l * pos_w[h][l]
__global__ void emb_bias_kernel(const float* __restrict__ lin_b,
                                const float* __restrict__ pos_w,
                                const float* __restrict__ pos_b,
                                float* __restrict__ out) {
  int h = blockIdx.x * blockDim.x + threadIdx.x;
  if (h >= HH_) return;
  float s = 0.0f;
  for (int l = 0; l < L_; ++l) s += (float)l * pos_w[(long long)h * L_ + l];
  out[h] = s + pos_b[h] + lin_b[h];
}

// cipad[b][0..1][:] = 1.0  (torch left-pad .fill_(1))
__global__ void cipad_init_kernel(float* __restrict__ cipad) {
  int i = blockIdx.x * blockDim.x + threadIdx.x;
  if (i >= B_ * 2 * H2_) return;
  int c = i % H2_;
  int r = i / H2_;
  cipad[((long long)(r >> 1) * LP_ + (r & 1)) * H2_ + c] = 1.0f;
}

// GLU: h[r][c] = y[r][c] * sigmoid(y[r][c+H2])
__global__ void glu_kernel(const float* __restrict__ y, float* __restrict__ h) {
  long long i = (long long)blockIdx.x * blockDim.x + threadIdx.x;
  if (i >= (long long)M_ * H2_) return;
  long long r = i >> 10;
  int c = (int)(i & (H2_ - 1));
  float a = y[r * H4_ + c];
  float g = y[r * H4_ + H2_ + c];
  h[i] = a * (1.0f / (1.0f + __expf(-g)));
}

// row softmax over ncols, one 128-thread block per row
__global__ __launch_bounds__(128)
void softmax_kernel(float* __restrict__ x, int ncols) {
  __shared__ float red[128];
  float* p = x + (long long)blockIdx.x * ncols;
  int t = threadIdx.x;
  float m = -3.4e38f;
  for (int c = t; c < ncols; c += 128) m = fmaxf(m, p[c]);
  red[t] = m; __syncthreads();
  for (int s = 64; s > 0; s >>= 1) { if (t < s) red[t] = fmaxf(red[t], red[t + s]); __syncthreads(); }
  m = red[0]; __syncthreads();
  float sum = 0.0f;
  for (int c = t; c < ncols; c += 128) { float e = __expf(p[c] - m); p[c] = e; sum += e; }
  red[t] = sum; __syncthreads();
  for (int s = 64; s > 0; s >>= 1) { if (t < s) red[t] += red[t + s]; __syncthreads(); }
  float inv = 1.0f / red[0];
  for (int c = t; c < ncols; c += 128) p[c] *= inv;
}

// cipad[b][l+2][:] += h[r][:]
__global__ void cipad_resid_kernel(float* __restrict__ cipad, const float* __restrict__ h) {
  long long i = (long long)blockIdx.x * blockDim.x + threadIdx.x;
  if (i >= (long long)M_ * H2_) return;
  long long r = i >> 10;
  int c = (int)(i & (H2_ - 1));
  long long b = r >> 9;
  long long l = r & (L_ - 1);
  cipad[(((b * LP_) + l + 2) << 10) + c] += h[i];
}

// ---------------------------------------------------------------------------
// Launch
// ---------------------------------------------------------------------------
static inline void gemm(hipStream_t s,
                        const float* A, long long a_off, int lda, long long a_be,
                        const _Float16* W, int ldw, long long w_bs,
                        const float* bias, const float* addsrc,
                        float* C, long long c_off, int ldc, long long c_be,
                        int M, int N, int K, int rpb) {
  if ((M % 128 == 0) && (N % 128 == 0) && (rpb % 128 == 0)) {
    dim3 grid(N / 128, M / 128);
    wmma_gemm_kernel<4, 4, 2, 2, true><<<grid, 128, 0, s>>>(
        A, a_off, lda, a_be, W, ldw, w_bs, bias, addsrc, C, c_off, ldc, c_be, M, N, K, rpb);
  } else {
    dim3 grid(N / 64, M / 64);
    wmma_gemm_kernel<2, 2, 2, 2, false><<<grid, 128, 0, s>>>(
        A, a_off, lda, a_be, W, ldw, w_bs, bias, addsrc, C, c_off, ldc, c_be, M, N, K, rpb);
  }
}

extern "C" void kernel_launch(void* const* d_in, const int* in_sizes, int n_in,
                              void* d_out, int out_size, void* d_ws, size_t ws_size,
                              hipStream_t stream) {
  const float* target   = (const float*)d_in[0];
  const float* he       = (const float*)d_in[1];
  const float* re       = (const float*)d_in[2];
  const float* lin_w    = (const float*)d_in[3];
  const float* lin_b    = (const float*)d_in[4];
  const float* pos_w    = (const float*)d_in[5];
  const float* pos_b    = (const float*)d_in[6];
  const float* tohid_w  = (const float*)d_in[7];
  const float* tohid_b  = (const float*)d_in[8];
  const float* conv_w   = (const float*)d_in[9];
  const float* conv_b   = (const float*)d_in[10];
  const float* attf_w   = (const float*)d_in[11];
  const float* attf_b   = (const float*)d_in[12];
  const float* attt_w   = (const float*)d_in[13];
  const float* attt_b   = (const float*)d_in[14];
  const float* fromh_w  = (const float*)d_in[15];
  const float* fromh_b  = (const float*)d_in[16];
  const float* out_w    = (const float*)d_in[17];
  const float* out_b    = (const float*)d_in[18];
  float* out = (float*)d_out;

  char* base = (char*)d_ws;
  size_t o = 0;
  auto alloc = [&](size_t bytes) -> char* {
    char* p = base + o;
    o = (o + bytes + 255) & ~(size_t)255;
    return p;
  };
  float*     emb     = (float*)alloc((size_t)M_ * HH_ * 4);
  float*     cipad   = (float*)alloc((size_t)B_ * LP_ * H2_ * 4);
  float*     ybuf    = (float*)alloc((size_t)M_ * H4_ * 4);
  float*     hbuf    = (float*)alloc((size_t)M_ * H2_ * 4);
  float*     rcbuf   = (float*)alloc((size_t)M_ * HH_ * 4);   // also ctx / hid
  float*     attn    = (float*)alloc((size_t)B_ * L_ * L_ * 4);
  float*     embBias = (float*)alloc((size_t)HH_ * 4);
  _Float16*  convW16 = (_Float16*)alloc((size_t)NLAYERS_ * H4_ * 3 * H2_ * 2);
  _Float16*  he16    = (_Float16*)alloc((size_t)B_ * L_ * HH_ * 2);
  _Float16*  reT16   = (_Float16*)alloc((size_t)B_ * HH_ * L_ * 2);
  _Float16*  lin16   = (_Float16*)alloc((size_t)HH_ * X_ * 2);
  _Float16*  tohid16 = (_Float16*)alloc((size_t)H2_ * HH_ * 2);
  _Float16*  attf16  = (_Float16*)alloc((size_t)HH_ * H2_ * 2);
  _Float16*  attt16  = (_Float16*)alloc((size_t)H2_ * HH_ * 2);
  _Float16*  fromh16 = (_Float16*)alloc((size_t)HH_ * H2_ * 2);
  _Float16*  outw16  = (_Float16*)alloc((size_t)X_ * HH_ * 2);
  (void)ws_size; (void)in_sizes; (void)n_in; (void)out_size;

  const int TPB = 256;
  auto nblk = [](long long n, int tpb) { return (unsigned)((n + tpb - 1) / tpb); };

  // ---- per-launch weight prep ---------------------------------------------
  cvt_f16_kernel<<<nblk((long long)HH_ * X_, TPB), TPB, 0, stream>>>(lin_w, lin16, (long long)HH_ * X_);
  cvt_f16_kernel<<<nblk((long long)H2_ * HH_, TPB), TPB, 0, stream>>>(tohid_w, tohid16, (long long)H2_ * HH_);
  cvt_f16_kernel<<<nblk((long long)HH_ * H2_, TPB), TPB, 0, stream>>>(attf_w, attf16, (long long)HH_ * H2_);
  cvt_f16_kernel<<<nblk((long long)H2_ * HH_, TPB), TPB, 0, stream>>>(attt_w, attt16, (long long)H2_ * HH_);
  cvt_f16_kernel<<<nblk((long long)HH_ * H2_, TPB), TPB, 0, stream>>>(fromh_w, fromh16, (long long)HH_ * H2_);
  cvt_f16_kernel<<<nblk((long long)X_ * HH_, TPB), TPB, 0, stream>>>(out_w, outw16, (long long)X_ * HH_);
  cvt_f16_kernel<<<nblk((long long)B_ * L_ * HH_, TPB), TPB, 0, stream>>>(he, he16, (long long)B_ * L_ * HH_);
  transpose_re_kernel<<<nblk((long long)B_ * HH_ * L_, TPB), TPB, 0, stream>>>(re, reT16);
  conv_repack_kernel<<<nblk((long long)NLAYERS_ * H4_ * 3 * H2_, TPB), TPB, 0, stream>>>(
      conv_w, convW16, (long long)NLAYERS_ * H4_ * 3 * H2_);
  emb_bias_kernel<<<1, HH_, 0, stream>>>(lin_b, pos_w, pos_b, embBias);
  cipad_init_kernel<<<nblk(B_ * 2 * H2_, TPB), TPB, 0, stream>>>(cipad);

  // ---- emb = target @ lin_w^T + (lin_b + pos) -----------------------------
  gemm(stream, target, 0, X_, 0, lin16, X_, 0, embBias, nullptr,
       emb, 0, HH_, 0, M_, HH_, X_, M_);

  // ---- ci0 = emb @ to_hidden_w^T + b -> cipad[b, l+2, :] ------------------
  gemm(stream, emb, 0, HH_, 0, tohid16, HH_, 0, tohid_b, nullptr,
       cipad, 2 * H2_, H2_, 2 * H2_, M_, H2_, HH_, L_);

  for (int l = 0; l < NLAYERS_; ++l) {
    // conv as GEMM: A-row r = 3072 contiguous floats at cipad[b, l_pos, 0]
    gemm(stream, cipad, 0, H2_, 2 * H2_,
         convW16 + (size_t)l * H4_ * 3 * H2_, 3 * H2_, 0,
         conv_b + (size_t)l * H4_, nullptr,
         ybuf, 0, H4_, 0, M_, H4_, 3 * H2_, L_);

    glu_kernel<<<nblk((long long)M_ * H2_, TPB), TPB, 0, stream>>>(ybuf, hbuf);

    // rc = h @ att_from_w^T + b + emb
    gemm(stream, hbuf, 0, H2_, 0, attf16, H2_, 0, attf_b, emb,
         rcbuf, 0, HH_, 0, M_, HH_, H2_, M_);

    // scores[b] = rc[b] @ he[b]^T
    gemm(stream, rcbuf, 0, HH_, 0, he16, HH_, (long long)L_ * HH_, nullptr, nullptr,
         attn, 0, L_, 0, M_, L_, HH_, L_);

    softmax_kernel<<<M_, 128, 0, stream>>>(attn, L_);

    // ctx[b] = attn[b] @ re[b]
    gemm(stream, attn, 0, L_, 0, reT16, L_, (long long)HH_ * L_, nullptr, nullptr,
         rcbuf, 0, HH_, 0, M_, HH_, L_, L_);

    // h = h + (ctx @ att_to_w^T + b)
    gemm(stream, rcbuf, 0, HH_, 0, attt16, HH_, 0, attt_b, hbuf,
         hbuf, 0, H2_, 0, M_, H2_, HH_, M_);

    // ci = h + ci
    cipad_resid_kernel<<<nblk((long long)M_ * H2_, TPB), TPB, 0, stream>>>(cipad, hbuf);
  }

  // ---- hid = ci @ from_hidden_w^T + b -------------------------------------
  gemm(stream, cipad, 2 * H2_, H2_, 2 * H2_, fromh16, H2_, 0, fromh_b, nullptr,
       rcbuf, 0, HH_, 0, M_, HH_, H2_, L_);

  // ---- out = hid @ out_w^T + b  (N=64 -> small tile path) -----------------
  gemm(stream, rcbuf, 0, HH_, 0, outw16, HH_, 0, out_b, nullptr,
       out, 0, X_, 0, M_, X_, HH_, M_);
}
